// WindowAttention_78082505441452
// MI455X (gfx1250) — matrix operand order repaired and checked by reading, not compiled, not run on hardware
//
#include <hip/hip_runtime.h>

// ---------------------------------------------------------------------------
// Swin shifted-window attention for MI455X (gfx1250, wave32, WMMA bf16).
// B=8, C=128, H=W=224, NH=4, DH=32, WS=7, T=49, shift=3, NWIN=1024.
// One workgroup (8 wave32, 256 thr) per window; each head owned by a wave pair.
// All LDS fragment traffic is ds_load_b128 / ds_store_b128; the shift-mask +
// relative-position bias is pre-folded into 4 window-class matrices bmT[cls].
// ---------------------------------------------------------------------------

typedef __attribute__((ext_vector_type(16))) __bf16 v16bf;
typedef __attribute__((ext_vector_type(8)))  float  v8f;

union FragBF { v16bf v; unsigned u[8]; uint4 q[2]; };

__device__ __forceinline__ unsigned short f2bf(float f) {
  unsigned u = __builtin_bit_cast(unsigned, f);
  u += 0x7FFFu + ((u >> 16) & 1u);          // round-to-nearest-even
  return (unsigned short)(u >> 16);
}
__device__ __forceinline__ unsigned pk2(float a, float b) {
  return (unsigned)f2bf(a) | ((unsigned)f2bf(b) << 16);
}

// 8 consecutive dwords (B-fragment pattern)
__device__ __forceinline__ FragBF ldFrag8(const unsigned* p) {
  FragBF f;
  f.q[0] = *(const uint4*)(p);
  f.q[1] = *(const uint4*)(p + 4);
  return f;
}
// two groups of 4 dwords at +0 and +8 (A-fragment pattern)
__device__ __forceinline__ FragBF ldFragSplit(const unsigned* p) {
  FragBF f;
  f.q[0] = *(const uint4*)(p);
  f.q[1] = *(const uint4*)(p + 8);
  return f;
}
__device__ __forceinline__ v8f bfwmma(const FragBF& a, const FragBF& b, v8f c) {
  return __builtin_amdgcn_wmma_f32_16x16x32_bf16(false, a.v, false, b.v,
                                                 (short)0, c, false, false);
}

#define NWIN_ 1024
#define SCALE_ 0.17677669529663687f  /* 32^-0.5 */

// ----------------------- prep kernel ---------------------------------------
// wbf: 4x(128x128) bf16 weights.  bmT[cls][s][t] (4x64x64 f32): relative-
// position bias with the cyclic-shift mask folded in as -1e10, transposed so
// the score epilogue reads contiguous t. cls = (wh==31)*2 + (ww==31).
__global__ void swin_prep_kernel(const float* __restrict__ table,
                                 const float* __restrict__ Wq,
                                 const float* __restrict__ Wk,
                                 const float* __restrict__ Wv,
                                 const float* __restrict__ Wo,
                                 unsigned short* __restrict__ wbf,
                                 float* __restrict__ bmT) {
  int idx = blockIdx.x * blockDim.x + threadIdx.x;
  int stride = gridDim.x * blockDim.x;
  const int nW = 128 * 128;
  for (int e = idx; e < 4 * nW; e += stride) {
    int g = e / nW, r = e % nW;
    const float* src = (g == 0) ? Wq : (g == 1) ? Wk : (g == 2) ? Wv : Wo;
    wbf[e] = f2bf(src[r]);
  }
  for (int e = idx; e < 4 * 64 * 64; e += stride) {
    const int cls = e >> 12;
    const int s = (e >> 6) & 63;
    const int t = e & 63;
    float v = 0.f;
    if (s < 49 && t < 49) {
      const int it = t / 7, jt = t % 7, is2 = s / 7, js = s % 7;
      const int eh = cls >> 1, ew = cls & 1;
      const int rt = (eh ? ((it  < 4) ? 1 : 2) : 0) * 3 + (ew ? ((jt < 4) ? 1 : 2) : 0);
      const int rs = (eh ? ((is2 < 4) ? 1 : 2) : 0) * 3 + (ew ? ((js < 4) ? 1 : 2) : 0);
      v = (rt != rs) ? -1e10f : table[(is2 - it + 6) * 13 + (js - jt + 6)];
    }
    bmT[e] = v;
  }
}

// transposed projection tile:  dst[t][d] = sum_c W[d][c] X[t][c]  (one d-tile)
__device__ __forceinline__ void projT(const unsigned* __restrict__ Wu,
                                      const float* __restrict__ bvec,
                                      const unsigned* __restrict__ Xu,
                                      unsigned* __restrict__ dstU,
                                      float scale, int mt, int lo16, int hi) {
  const v8f vz = {0.f,0.f,0.f,0.f,0.f,0.f,0.f,0.f};
  const int dbase = mt * 16 + hi * 8;
  float bb[8];
#pragma unroll
  for (int r = 0; r < 8; ++r) bb[r] = bvec[dbase + r] * scale;
  FragBF Af[4];
  const int drow = mt * 16 + lo16;
#pragma unroll
  for (int kit = 0; kit < 4; ++kit)
    Af[kit] = ldFrag8(Wu + drow * 64 + kit * 16 + hi * 8);
#pragma unroll
  for (int nt = 0; nt < 4; ++nt) {
    const int t = nt * 16 + lo16;
    v8f acc = vz;
#pragma unroll
    for (int kit = 0; kit < 4; ++kit) {
      FragBF Bf = ldFrag8(Xu + t * 64 + kit * 16 + hi * 8);
      acc = bfwmma(Af[kit], Bf, acc);
    }
    uint4 o = make_uint4(pk2(acc[0]*scale + bb[0], acc[1]*scale + bb[1]),
                         pk2(acc[2]*scale + bb[2], acc[3]*scale + bb[3]),
                         pk2(acc[4]*scale + bb[4], acc[5]*scale + bb[5]),
                         pk2(acc[6]*scale + bb[6], acc[7]*scale + bb[7]));
    *(uint4*)(dstU + t * 64 + dbase / 2) = o;
  }
}

// ----------------------- main kernel ---------------------------------------
// Dynamic LDS layout (bytes):
//   [0,16384)        Xbf   u16[64][128]   token-major, rows 49..63 = 0
//   [16384,32768)    Qbf   u16[64][128]   token-major, pre-scaled by SCALE
//   [32768,49152)    Kbf   u16[64][128]   token-major
//   [49152,65536)    Vt    u16[128][64]   d-major (transposed V)
//   [65536,131072)   per-head 16KB: Sc f32[64][64]; P row t (64 bf16) embedded
//                    in the first 128B of Sc row t (row stride 64 dwords)
//   [131072,147456)  Obf   u16[64][128]   token-major
//   Yf f32[64][128] aliases [0,32768) after the post-AV barrier.
__global__ __launch_bounds__(256)
void swin_attn_kernel(const float* __restrict__ x,
                      const float* __restrict__ bq, const float* __restrict__ bk,
                      const float* __restrict__ bv, const float* __restrict__ bo,
                      const unsigned* __restrict__ wbf,
                      const float* __restrict__ bmT,
                      float* __restrict__ out) {
  extern __shared__ char smem[];
  unsigned*       XbfU = (unsigned*)(smem);
  unsigned*       QbfU = (unsigned*)(smem + 16384);
  unsigned*       KbfU = (unsigned*)(smem + 32768);
  unsigned*       VtU  = (unsigned*)(smem + 49152);
  unsigned short* Obf  = (unsigned short*)(smem + 131072);
  float*          Yf   = (float*)(smem);

  const int tid  = threadIdx.x;
  const int lane = tid & 31;
  const int wave = tid >> 5;           // 0..7
  const int lo16 = lane & 15;
  const int hi   = (lane >= 16) ? 1 : 0;

  const int blk = blockIdx.x;
  const int b   = blk >> 10;
  const int win = blk & (NWIN_ - 1);
  const int wh  = win >> 5;
  const int ww  = win & 31;
  const int baseh = wh * 7 + 3;        // input cyclic shift -3 == +3 read offset
  const int basew = ww * 7 + 3;
  const int cls = ((wh == 31) ? 2 : 0) + ((ww == 31) ? 1 : 0);

  const v8f vz = {0.f,0.f,0.f,0.f,0.f,0.f,0.f,0.f};

  // ---- Phase 1: gather shifted window tile -> LDS bf16 (channel-pair packed)
  for (int e = tid; e < 64 * 64; e += 256) {
    const int t  = e & 63;             // consecutive lanes -> consecutive w
    const int cp = e >> 6;             // channel pair
    unsigned dw = 0;
    if (t < 49) {
      int h = baseh + t / 7; if (h >= 224) h -= 224;
      int w = basew + t % 7; if (w >= 224) w -= 224;
      const float* px = x + (((b * 128 + 2 * cp) * 224 + h) * 224) + w;
      dw = pk2(px[0], px[224 * 224]);
    }
    XbfU[t * 64 + cp] = dw;
  }
  __syncthreads();

  // ---- Phase 2: projections; each wave owns one 16-wide d-tile
  projT(wbf + 0 * 8192, bq, XbfU, QbfU, SCALE_, wave, lo16, hi);  // Q (scaled)
  projT(wbf + 1 * 8192, bk, XbfU, KbfU, 1.0f,   wave, lo16, hi);  // K
  {  // V = X Wv^T, original orientation -> contiguous stores into Vt[d][t]
    const unsigned* WvU = wbf + 2 * 8192;
    const int d = wave * 16 + lo16;
    FragBF Bf[4];
#pragma unroll
    for (int kit = 0; kit < 4; ++kit)
      Bf[kit] = ldFrag8(WvU + d * 64 + kit * 16 + hi * 8);
    const float bias = bv[d];
#pragma unroll
    for (int mt = 0; mt < 4; ++mt) {
      const int m = mt * 16 + lo16;
      v8f acc = vz;
#pragma unroll
      for (int kit = 0; kit < 4; ++kit) {
        FragBF Af = ldFrag8(XbfU + m * 64 + kit * 16 + hi * 8);
        acc = bfwmma(Af, Bf[kit], acc);
      }
      const int tbase = mt * 16 + hi * 8;
      uint4 o = make_uint4(pk2(acc[0]+bias, acc[1]+bias),
                           pk2(acc[2]+bias, acc[3]+bias),
                           pk2(acc[4]+bias, acc[5]+bias),
                           pk2(acc[6]+bias, acc[7]+bias));
      *(uint4*)(VtU + d * 32 + tbase / 2) = o;
    }
  }
  __syncthreads();

  // ---- Phase 3: attention; head h = wave>>1, wave pair splits query rows
  const int h   = wave >> 1;
  const int mth = wave & 1;            // which half of the query rows
  float*    Sc  = (float*)(smem + 65536 + h * 16384);   // f32[64][64]
  unsigned* Pu  = (unsigned*)Sc;       // P row t = first 32 dwords of Sc row t
  {
    // scores = (Q*SCALE) K^T + bmT[cls]  (bias & shift mask pre-folded)
    const float* bmBase = bmT + cls * 4096;
#pragma unroll
    for (int nt = 0; nt < 4; ++nt) {
      const int s = nt * 16 + lo16;
      FragBF Bf = ldFrag8(KbfU + s * 64 + h * 16 + hi * 8);
#pragma unroll
      for (int i = 0; i < 2; ++i) {
        const int mt = 2 * mth + i;
        const int tq = mt * 16 + lo16;
        FragBF Af = ldFrag8(QbfU + tq * 64 + h * 16 + hi * 8);
        v8f acc = vz;
        acc = bfwmma(Af, Bf, acc);
        const int mbase = mt * 16 + hi * 8;
        const float4* bp = (const float4*)(bmBase + s * 64 + mbase);
        const float4 f0 = bp[0], f1 = bp[1];
        Sc[(mbase+0) * 64 + s] = acc[0] + f0.x;
        Sc[(mbase+1) * 64 + s] = acc[1] + f0.y;
        Sc[(mbase+2) * 64 + s] = acc[2] + f0.z;
        Sc[(mbase+3) * 64 + s] = acc[3] + f0.w;
        Sc[(mbase+4) * 64 + s] = acc[4] + f1.x;
        Sc[(mbase+5) * 64 + s] = acc[5] + f1.y;
        Sc[(mbase+6) * 64 + s] = acc[6] + f1.z;
        Sc[(mbase+7) * 64 + s] = acc[7] + f1.w;
      }
    }
    // softmax: one row per lane, each wave owns the rows it just wrote.
    // P written into the first 128B of the same Sc row -> no cross-wave alias.
    const int t = mth * 32 + lane;
    if (t < 49) {
      float v[49];
      const float4* row4 = (const float4*)(Sc + t * 64);
#pragma unroll
      for (int q = 0; q < 12; ++q) {
        float4 f = row4[q];
        v[4*q+0] = f.x; v[4*q+1] = f.y; v[4*q+2] = f.z; v[4*q+3] = f.w;
      }
      v[48] = Sc[t * 64 + 48];
      float mx = -3.0e38f;
#pragma unroll
      for (int s = 0; s < 49; ++s) mx = fmaxf(mx, v[s]);
      float sum = 0.f;
#pragma unroll
      for (int s = 0; s < 49; ++s) { float e = __expf(v[s] - mx); v[s] = e; sum += e; }
      const float inv = 1.f / sum;
      unsigned pr[32];
#pragma unroll
      for (int sp = 0; sp < 24; ++sp) pr[sp] = pk2(v[2*sp] * inv, v[2*sp+1] * inv);
      pr[24] = (unsigned)f2bf(v[48] * inv);      // pad key 49 = 0
#pragma unroll
      for (int sp = 25; sp < 32; ++sp) pr[sp] = 0;
      uint4* prow = (uint4*)(Pu + t * 64);
#pragma unroll
      for (int q = 0; q < 8; ++q)
        prow[q] = make_uint4(pr[4*q], pr[4*q+1], pr[4*q+2], pr[4*q+3]);
    }
  }
  __syncthreads();   // AV reads all P rows of the head (written by wave pair)

  // ---- Phase 4: O^T = V^T P^T ; wave pair splits the head's 32 d-values
  {
    unsigned* Ou = (unsigned*)Obf;
    FragBF Af[2];
    const int drow = h * 32 + mth * 16 + lo16;
#pragma unroll
    for (int kit = 0; kit < 2; ++kit)
      Af[kit] = ldFrag8(VtU + drow * 32 + kit * 16 + hi * 8);
    const int dbase = h * 32 + mth * 16 + hi * 8;
#pragma unroll
    for (int nt = 0; nt < 4; ++nt) {
      const int t = nt * 16 + lo16;
      v8f acc = vz;
#pragma unroll
      for (int kit = 0; kit < 2; ++kit) {
        FragBF Bf = ldFrag8(Pu + t * 64 + kit * 16 + hi * 8);  // row stride 64
        acc = bfwmma(Af[kit], Bf, acc);
      }
      uint4 o = make_uint4(pk2(acc[0], acc[1]), pk2(acc[2], acc[3]),
                           pk2(acc[4], acc[5]), pk2(acc[6], acc[7]));
      *(uint4*)(Ou + t * 64 + dbase / 2) = o;
    }
  }
  __syncthreads();

  // ---- Phase 5: Y^T = Wo O^T + bo -> Yf fp32 (aliases X/Q); one c-tile/wave
  {
    const unsigned* Ou  = (const unsigned*)Obf;
    const unsigned* WoU = wbf + 3 * 8192;
    const int crow = wave * 16 + lo16;
    FragBF Af[4];
#pragma unroll
    for (int kit = 0; kit < 4; ++kit)
      Af[kit] = ldFrag8(WoU + crow * 64 + kit * 16 + hi * 8);
    const int cbase = wave * 16 + hi * 8;
    float bb[8];
#pragma unroll
    for (int r = 0; r < 8; ++r) bb[r] = bo[cbase + r];
#pragma unroll
    for (int nt = 0; nt < 4; ++nt) {
      const int t = nt * 16 + lo16;
      v8f acc = vz;
#pragma unroll
      for (int kit = 0; kit < 4; ++kit) {
        FragBF Bf = ldFrag8(Ou + t * 64 + kit * 16 + hi * 8);
        acc = bfwmma(Af[kit], Bf, acc);
      }
      *(float4*)(Yf + t * 128 + cbase) =
          make_float4(acc[0]+bb[0], acc[1]+bb[1], acc[2]+bb[2], acc[3]+bb[3]);
      *(float4*)(Yf + t * 128 + cbase + 4) =
          make_float4(acc[4]+bb[4], acc[5]+bb[5], acc[6]+bb[6], acc[7]+bb[7]);
    }
  }
  __syncthreads();

  // ---- Phase 6: un-partition + roll(+3,+3) scatter to global
  for (int e = tid; e < 49 * 128; e += 256) {
    const int t = e % 49;              // consecutive lanes -> consecutive w
    const int c = e / 49;
    int hh = baseh + t / 7; if (hh >= 224) hh -= 224;
    int wc = basew + t % 7; if (wc >= 224) wc -= 224;
    out[(((b * 128 + c) * 224 + hh) * 224) + wc] = Yf[t * 128 + c];
  }
}

// ----------------------- launcher ------------------------------------------
extern "C" void kernel_launch(void* const* d_in, const int* in_sizes, int n_in,
                              void* d_out, int out_size, void* d_ws, size_t ws_size,
                              hipStream_t stream) {
  const float* x     = (const float*)d_in[0];
  const float* table = (const float*)d_in[1];
  const float* Wq    = (const float*)d_in[2];
  const float* bq    = (const float*)d_in[3];
  const float* Wk    = (const float*)d_in[4];
  const float* bk    = (const float*)d_in[5];
  const float* Wv    = (const float*)d_in[6];
  const float* bv    = (const float*)d_in[7];
  const float* Wo    = (const float*)d_in[8];
  const float* bo    = (const float*)d_in[9];
  float* out = (float*)d_out;

  unsigned short* wbf = (unsigned short*)d_ws;                 // 131072 B
  float* bmT = (float*)((char*)d_ws + 4 * 128 * 128 * sizeof(unsigned short));

  swin_prep_kernel<<<64, 256, 0, stream>>>(table, Wq, Wk, Wv, Wo, wbf, bmT);

  const size_t ldsBytes = 147456;  // 144 KB dynamic LDS (320 KB/WGP on CDNA5)
  swin_attn_kernel<<<8 * NWIN_, 256, ldsBytes, stream>>>(
      x, bq, bk, bv, bo, (const unsigned*)wbf, bmT, out);
}
